// TopoformerLayer_58755152609840
// MI455X (gfx1250) — matrix operand order repaired and checked by reading, not compiled, not run on hardware
//
#include <hip/hip_runtime.h>
#include <hip/hip_bf16.h>
#include <math.h>

// ---------------------------------------------------------------------------
// Topoformer exact-kNN: fused bf16-WMMA Gram + d^2 top-32 (sqrt deferred).
//   B=8, S=4096, D=256, k=32.  Outputs (float*): [B*S*32] neighbor indices
//   (as float), then [B*S*32] distances, ascending per row.
// ---------------------------------------------------------------------------

typedef __attribute__((ext_vector_type(16))) __bf16 v16bf;
typedef __attribute__((ext_vector_type(8)))  __bf16 bf16x8;
typedef __attribute__((ext_vector_type(4)))  __bf16 bf16x4;
typedef __attribute__((ext_vector_type(8)))  float  v8f;
typedef __attribute__((ext_vector_type(4)))  float  f32x4;

#define BB 8
#define SS 4096
#define DD 256
#define KK 32
#define NWAVES 4
#define NTILES (SS / 16)                 // 256 column tiles of 16
#define TILES_PER_WAVE (NTILES / NWAVES) // 64

// ---------- fp32 -> bf16 conversion of the embedding tensor (one pass) ------
__global__ void convert_bf16_kernel(const float* __restrict__ in,
                                    __bf16* __restrict__ out, int n4) {
  int i = blockIdx.x * blockDim.x + threadIdx.x;
  if (i >= n4) return;
  f32x4 v = ((const f32x4*)in)[i];
  bf16x4 o;
  o.x = (__bf16)v.x; o.y = (__bf16)v.y; o.z = (__bf16)v.z; o.w = (__bf16)v.w;
  ((bf16x4*)out)[i] = o;
}

// ---------- exact fp32 row norms -------------------------------------------
__global__ void rownorm_kernel(const float* __restrict__ emb,
                               float* __restrict__ sq, int rows) {
  int r = blockIdx.x * blockDim.x + threadIdx.x;
  if (r >= rows) return;
  const f32x4* p = (const f32x4*)(emb + (size_t)r * DD);
  float acc = 0.f;
#pragma unroll
  for (int i = 0; i < DD / 4; ++i) {
    f32x4 v = p[i];
    acc += v.x * v.x + v.y * v.y + v.z * v.z + v.w * v.w;
  }
  sq[r] = acc;
}

// ---------- fused Gram(WMMA) + d^2 top-32 ----------------------------------
// Block = 128 threads (4 waves, wave32). Block owns 16 rows of one batch.
// Wave w owns column tiles [w*64, w*64+64). A tile resident in VGPRs.
__global__ __launch_bounds__(128)
void topk_wmma_kernel(const __bf16* __restrict__ ebf,
                      const float* __restrict__ sq,
                      float* __restrict__ outN,
                      float* __restrict__ outD) {
  __shared__ float tile[NWAVES][16][17];  // per-wave d^2 staging (pad 17)
  __shared__ float topv[NWAVES][16][33];  // per-wave per-row top-32 d^2
  __shared__ int   topi[NWAVES][16][33];  // matching column indices
  __shared__ float rowsq[16];

  const int tid  = threadIdx.x;
  const int w    = tid >> 5;        // wave id (wave32)
  const int lane = tid & 31;
  const int h    = lane >> 4;       // lane half (K-group select)
  const int n    = lane & 15;       // row/col slot within tile

  const int blocksPerBatch = SS / 16;
  const int b       = blockIdx.x / blocksPerBatch;
  const int rowBase = (blockIdx.x % blocksPerBatch) * 16;

  // init per-wave top lists to +inf
  for (int i = tid; i < NWAVES * 16 * 32; i += 128) {
    int ww = i >> 9, rr = (i >> 5) & 15, ss = i & 31;
    topv[ww][rr][ss] = __builtin_inff();
    topi[ww][rr][ss] = 0;
  }
  if (tid < 16) rowsq[tid] = sq[b * SS + rowBase + tid];
  __syncthreads();

  // -------- resident A fragments: 16 rows x 256 dims, bf16 ------------------
  // 16-bit A 16x32 layout: lane half h holds K = 8h+0..7 (elems 0..7) and
  // K = 16+8h+0..7 (elems 8..15).
  const __bf16* arow = ebf + ((size_t)b * SS + rowBase + n) * DD;
  v16bf afrag[8];
#pragma unroll
  for (int kk = 0; kk < 8; ++kk) {
    union { v16bf v; bf16x8 half8[2]; } u;
    u.half8[0] = *(const bf16x8*)(arow + kk * 32 + 8 * h);
    u.half8[1] = *(const bf16x8*)(arow + kk * 32 + 16 + 8 * h);
    afrag[kk] = u.v;
  }

  float curmax = __builtin_inff();  // running 32nd-smallest d^2 for my row
  int   amax   = 0;

  const __bf16* bbase = ebf + (size_t)b * SS * DD;
  const float*  sqb   = sq + b * SS;

  for (int t = 0; t < TILES_PER_WAVE; ++t) {
    const int nt      = w * TILES_PER_WAVE + t;
    const int colBase = nt * 16;
    const __bf16* brow = bbase + (size_t)(colBase + n) * DD;

    // prefetch the whole next-tile row for this lane (512 B = 4 cachelines)
    if (t + 1 < TILES_PER_WAVE) {
      const char* pf = (const char*)(brow + 16 * DD);
      __builtin_prefetch(pf + 0,   0, 1);
      __builtin_prefetch(pf + 128, 0, 1);
      __builtin_prefetch(pf + 256, 0, 1);
      __builtin_prefetch(pf + 384, 0, 1);
    }

    // 16-bit B 32x16 layout: lane = column, half h holds K = 8h+0..7 and 16+8h+0..7.
    // Load ALL 8 fragments first (one clause) so WMMAs overlap in-flight loads.
    v16bf bfrag[8];
#pragma unroll
    for (int kk = 0; kk < 8; ++kk) {
      union { v16bf v; bf16x8 half8[2]; } ub;
      ub.half8[0] = *(const bf16x8*)(brow + kk * 32 + 8 * h);
      ub.half8[1] = *(const bf16x8*)(brow + kk * 32 + 16 + 8 * h);
      bfrag[kk] = ub.v;
    }

    v8f acc;
#pragma unroll
    for (int j = 0; j < 8; ++j) acc[j] = 0.f;
#pragma unroll
    for (int kk = 0; kk < 8; ++kk) {
      acc = __builtin_amdgcn_wmma_f32_16x16x32_bf16(
          /*neg_a=*/false, afrag[kk], /*neg_b=*/false, bfrag[kk],
          /*c_mod=*/(short)0, acc, /*reuse_a=*/false, /*reuse_b=*/false);
    }

    // C layout: lane holds column n; VGPR j holds row m = j + 8h.
    // Stay in the d^2 domain: sqrt is monotonic -> identical top-k.
    const int   col   = colBase + n;
    const float colsq = sqb[col];
#pragma unroll
    for (int j = 0; j < 8; ++j) {
      const int m  = j + 8 * h;
      float d2 = rowsq[m] + colsq - 2.0f * acc[j];
      if (rowBase + m == col) d2 = __builtin_inff();  // diagonal -> inf
      tile[w][m][n] = d2;
    }
    // intra-wave LDS handoff: LDS ops are in-order per wave; fence compiler
    asm volatile("s_wait_dscnt 0" ::: "memory");

    // lanes 0..15: threshold-filtered insertion into this wave's row list
    if (h == 0) {
      const int r = n;
#pragma unroll 1
      for (int c = 0; c < 16; ++c) {
        float v = tile[w][r][c];
        if (v < curmax) {
          topv[w][r][amax] = v;
          topi[w][r][amax] = colBase + c;
          float mx = -__builtin_inff(); int mi = 0;
#pragma unroll 1
          for (int s = 0; s < 32; ++s) {
            float tv = topv[w][r][s];
            if (tv > mx) { mx = tv; mi = s; }
          }
          curmax = mx; amax = mi;
        }
      }
    }
  }

  __syncthreads();

  // -------- merge 4 partitions (4x32 candidates) -> 32 ascending ------------
  // sqrt applied only here (32 values per row), single-instruction v_sqrt_f32.
  if (tid < 16) {
    const int r = tid;
    const size_t orow = ((size_t)b * SS + rowBase + r) * KK;
#pragma unroll 1
    for (int j = 0; j < KK; ++j) {
      float best = __builtin_inff(); int bw = 0, bs = 0;
#pragma unroll 1
      for (int ww = 0; ww < NWAVES; ++ww)
        for (int s = 0; s < 32; ++s) {
          float v = topv[ww][r][s];
          if (v < best) { best = v; bw = ww; bs = s; }
        }
      outN[orow + j] = (float)topi[bw][r][bs];                 // index as float
      outD[orow + j] = __builtin_amdgcn_sqrtf(fmaxf(best, 1e-12f));
      topv[bw][r][bs] = __builtin_inff();
    }
  }
}

// ---------------------------------------------------------------------------
extern "C" void kernel_launch(void* const* d_in, const int* in_sizes, int n_in,
                              void* d_out, int out_size, void* d_ws, size_t ws_size,
                              hipStream_t stream) {
  const float* emb = (const float*)d_in[0];
  // d_in[1] = projections: hash codes are not part of the returned outputs.
  // d_in[2] = k (fixed 32).

  const size_t ebf_bytes = (size_t)BB * SS * DD * sizeof(__bf16);  // 16 MB
  __bf16* ebf = (__bf16*)d_ws;
  float*  sqn = (float*)((char*)d_ws + ebf_bytes);                 // 128 KB
  if (ws_size < ebf_bytes + (size_t)BB * SS * sizeof(float)) return;

  float* outN = (float*)d_out;                       // [B*S*32] indices
  float* outD = outN + (size_t)BB * SS * KK;         // [B*S*32] distances

  {
    int n4 = BB * SS * DD / 4;
    convert_bf16_kernel<<<(n4 + 255) / 256, 256, 0, stream>>>(emb, ebf, n4);
  }
  {
    int rows = BB * SS;
    rownorm_kernel<<<(rows + 255) / 256, 256, 0, stream>>>(emb, sqn, rows);
  }
  topk_wmma_kernel<<<BB * (SS / 16), 128, 0, stream>>>(ebf, sqn, outN, outD);
}
